// PAM_Module_4217657885008
// MI455X (gfx1250) — matrix operand order repaired and checked by reading, not compiled, not run on hardware
//
#include <hip/hip_runtime.h>
#include <hip/hip_bf16.h>

typedef __attribute__((ext_vector_type(16))) _Float16 v16h;
typedef __attribute__((ext_vector_type(8)))  _Float16 v8h;
typedef __attribute__((ext_vector_type(4)))  _Float16 v4h;
typedef __attribute__((ext_vector_type(8)))  float    v8f;
typedef __attribute__((ext_vector_type(4)))  float    v4f;
typedef __attribute__((ext_vector_type(4)))  int      v4i;

#if __has_builtin(__builtin_amdgcn_global_load_async_to_lds_b128) && \
    __has_builtin(__builtin_amdgcn_s_wait_asynccnt)
#define HAS_ASYNC_LDS 1
typedef __attribute__((address_space(1))) v4i* as1_v4i;
typedef __attribute__((address_space(3))) v4i* as3_v4i;
#else
#define HAS_ASYNC_LDS 0
#endif

// LDS row stride in halfs: 32 data + 8 pad -> 80B rows, 16B aligned, bank-staggered.
constexpr int LDST = 40;

// A fragment, 16x32 f16 (M x K). Lane l: row = l&15.
// elems 0..7  = A[row][ (l>>4)*8 + 0..7 ],  elems 8..15 = A[row][ 16 + (l>>4)*8 + 0..7 ]
__device__ __forceinline__ v16h frag_a(const _Float16* tile, int lane) {
  const int row = lane & 15, s = (lane >> 4) & 1;
  const _Float16* p = tile + row * LDST + s * 8;
  v8h lo = *(const v8h*)(p);
  v8h hi = *(const v8h*)(p + 16);
  v16h r;
#pragma unroll
  for (int i = 0; i < 8; ++i) { r[i] = lo[i]; r[8 + i] = hi[i]; }
  return r;
}

// B fragment, 32x16 f16 (K x N) from Bt tile stored [n][k] K-contig.
// Lane l: col = l&15; elems 0..7 = K (l>>4)*8 + 0..7, elems 8..15 = K 16 + (l>>4)*8 + 0..7.
__device__ __forceinline__ v16h frag_b(const _Float16* tile, int lane) {
  const int col = lane & 15, s = (lane >> 4) & 1;
  const _Float16* p = tile + col * LDST + s * 8;
  v8h lo = *(const v8h*)(p);
  v8h hi = *(const v8h*)(p + 16);
  v16h r;
#pragma unroll
  for (int i = 0; i < 8; ++i) { r[i] = lo[i]; r[8 + i] = hi[i]; }
  return r;
}

// Async (ASYNCcnt) copy of an R x 32-half K-contiguous f16 tile into LDS, 16B chunks.
template <int R>
__device__ __forceinline__ void stage_async_f16(_Float16* dst, const _Float16* src,
                                                int row0, int ld, int k0, int tid) {
#if HAS_ASYNC_LDS
  for (int idx = tid; idx < R * 4; idx += 256) {
    const int r = idx >> 2, c = idx & 3;
    const _Float16* g = src + (long long)(row0 + r) * ld + k0 + c * 8;
    _Float16* l = dst + r * LDST + c * 8;
    __builtin_amdgcn_global_load_async_to_lds_b128((as1_v4i)g, (as3_v4i)l, 0, 0);
  }
#endif
}

// Synchronous vectorized staging of an R x 32 tile (f16 16B chunks / f32 4-wide+convert).
template <int R, typename SrcT>
__device__ __forceinline__ void stage_sync(_Float16* dst, const SrcT* src,
                                           int row0, int ld, int k0, int tid) {
  if constexpr (sizeof(SrcT) == 2) {
    for (int idx = tid; idx < R * 4; idx += 256) {
      const int r = idx >> 2, c = idx & 3;
      *(v8h*)(dst + r * LDST + c * 8) =
          *(const v8h*)((const _Float16*)src + (long long)(row0 + r) * ld + k0 + c * 8);
    }
  } else {
    for (int idx = tid; idx < R * 8; idx += 256) {
      const int r = idx >> 3, c = idx & 7;
      v4f t = *(const v4f*)((const float*)src + (long long)(row0 + r) * ld + k0 + c * 4);
      v4h h;
#pragma unroll
      for (int i = 0; i < 4; ++i) h[i] = (_Float16)t[i];
      *(v4h*)(dst + r * LDST + c * 4) = h;
    }
  }
}

// Transpose-on-stage: source stored K-major (K x Rdim), gather columns (f32 activations).
template <int R, typename SrcT>
__device__ __forceinline__ void stage_trans(_Float16* dst, const SrcT* src,
                                            int row0, int ld, int k0, int tid) {
  constexpr int KPP = 256 / R;
  const int r = tid % R, kk = tid / R;
#pragma unroll
  for (int k = kk; k < 32; k += KPP)
    dst[r * LDST + k] = (_Float16)src[(long long)(k0 + k) * ld + (row0 + r)];
}

// Generic batched TN GEMM: D(MxNn) = A(MxK) * B(KxNn), B supplied as Bt (Nn x K).
// TA/TB: operand source stored K-major -> transpose while staging.
// BIAS: 0 none, 1 bias[col], 2 bias[row].
template <int BM, int BN, int WM, int WN, bool TA, bool TB, typename SrcT,
          bool OUT_F16, int BIAS>
__global__ __launch_bounds__(256) void gemm_wmma_k(
    const void* __restrict__ Ap, const void* __restrict__ Bp,
    const float* __restrict__ bias, void* __restrict__ Dp,
    int K, int lda, int ldb, int ldd,
    long long sA, long long sB, long long sD) {
  constexpr int AF = WM / 16, BF = WN / 16;
  constexpr int MW = BM / WM;
  static_assert((BM / WM) * (BN / WN) == 8, "block must be 8 waves");
  constexpr bool UA = HAS_ASYNC_LDS && !TA && !TB && sizeof(SrcT) == 2;
  constexpr int NBUF = UA ? 2 : 1;
  __shared__ alignas(16) _Float16 ldsA[NBUF * BM * LDST];
  __shared__ alignas(16) _Float16 ldsB[NBUF * BN * LDST];

  const int tid  = threadIdx.x;
  const int lane = tid & 31;
  const int w    = tid >> 5;
  const int wm   = (w % MW) * WM;
  const int wn   = (w / MW) * WN;
  const int m0   = blockIdx.y * BM;
  const int n0   = blockIdx.x * BN;

  const SrcT* A  = (const SrcT*)Ap + (long long)blockIdx.z * sA;
  const SrcT* Bt = (const SrcT*)Bp + (long long)blockIdx.z * sB;

  v8f acc[AF][BF];
  v8f vzero = {};
#pragma unroll
  for (int i = 0; i < AF; ++i)
#pragma unroll
    for (int j = 0; j < BF; ++j) acc[i][j] = vzero;

  int cur = 0;
  if constexpr (UA) {
    stage_async_f16<BM>(ldsA, (const _Float16*)A, m0, lda, 0, tid);
    stage_async_f16<BN>(ldsB, (const _Float16*)Bt, n0, ldb, 0, tid);
#if HAS_ASYNC_LDS
    __builtin_amdgcn_s_wait_asynccnt(0);
#endif
    __syncthreads();
  }

  for (int k0 = 0; k0 < K; k0 += 32) {
    if constexpr (UA) {
      // double-buffered: prefetch next tile asynchronously while computing this one
      const int nxt = cur ^ 1;
      if (k0 + 32 < K) {
        stage_async_f16<BM>(ldsA + nxt * BM * LDST, (const _Float16*)A, m0, lda, k0 + 32, tid);
        stage_async_f16<BN>(ldsB + nxt * BN * LDST, (const _Float16*)Bt, n0, ldb, k0 + 32, tid);
      }
    } else {
      if constexpr (TA) stage_trans<BM>(ldsA, A, m0, lda, k0, tid);
      else              stage_sync<BM>(ldsA, A, m0, lda, k0, tid);
      if constexpr (TB) stage_trans<BN>(ldsB, Bt, n0, ldb, k0, tid);
      else              stage_sync<BN>(ldsB, Bt, n0, ldb, k0, tid);
      __syncthreads();
    }

    const _Float16* tA = ldsA + cur * BM * LDST;
    const _Float16* tB = ldsB + cur * BN * LDST;
    v16h af[AF], bf[BF];
#pragma unroll
    for (int i = 0; i < AF; ++i) af[i] = frag_a(tA + (wm + i * 16) * LDST, lane);
#pragma unroll
    for (int j = 0; j < BF; ++j) bf[j] = frag_b(tB + (wn + j * 16) * LDST, lane);
#pragma unroll
    for (int i = 0; i < AF; ++i)
#pragma unroll
      for (int j = 0; j < BF; ++j)
        acc[i][j] = __builtin_amdgcn_wmma_f32_16x16x32_f16(
            false, af[i], false, bf[j], (short)0, acc[i][j], false, false);

    if constexpr (UA) {
#if HAS_ASYNC_LDS
      __builtin_amdgcn_s_wait_asynccnt(0);
#endif
      __syncthreads();
      cur ^= 1;
    } else {
      __syncthreads();
    }
  }

  // Epilogue: C/D layout — VGPR r: row = r + (lane&16 ? 8 : 0), col = lane&15.
  const int rsel = (lane & 16) ? 8 : 0;
  const int cl   = lane & 15;
  const long long dbase = (long long)blockIdx.z * sD;
#pragma unroll
  for (int i = 0; i < AF; ++i) {
#pragma unroll
    for (int j = 0; j < BF; ++j) {
      const int col = n0 + wn + j * 16 + cl;
      float bn_ = 0.0f;
      if constexpr (BIAS == 1) bn_ = bias[col];
#pragma unroll
      for (int r = 0; r < 8; ++r) {
        const int row = m0 + wm + i * 16 + rsel + r;
        float v = acc[i][j][r] + bn_;
        if constexpr (BIAS == 2) v += bias[row];
        const long long idx = dbase + (long long)row * ldd + col;
        if constexpr (OUT_F16)
          ((_Float16*)Dp)[idx] = (_Float16)v;
        else
          ((float*)Dp)[idx] = v;
      }
    }
  }
}

// In-place row softmax over f16 rows of length 4096 (one 256-thread block per row).
__global__ __launch_bounds__(256) void softmax_rows_k(_Float16* __restrict__ E) {
  __shared__ float red[256];
  _Float16* p = E + (long long)blockIdx.x * 4096;
  const int t = threadIdx.x;
  float v[16];
  float mx = -3.0e38f;
#pragma unroll
  for (int i = 0; i < 16; ++i) {
    v[i] = (float)p[t + (i << 8)];
    mx = fmaxf(mx, v[i]);
  }
  red[t] = mx;
  __syncthreads();
  for (int s = 128; s > 0; s >>= 1) {
    if (t < s) red[t] = fmaxf(red[t], red[t + s]);
    __syncthreads();
  }
  mx = red[0];
  __syncthreads();
  float sum = 0.0f;
#pragma unroll
  for (int i = 0; i < 16; ++i) {
    v[i] = __expf(v[i] - mx);
    sum += v[i];
  }
  red[t] = sum;
  __syncthreads();
  for (int s = 128; s > 0; s >>= 1) {
    if (t < s) red[t] += red[t + s];
    __syncthreads();
  }
  const float inv = 1.0f / red[0];
#pragma unroll
  for (int i = 0; i < 16; ++i) p[t + (i << 8)] = (_Float16)(v[i] * inv);
}

// result[b][c][n] = gamma * outT[b][n][c] + x[b][c][n]  (32x32 LDS transpose tiles)
__global__ __launch_bounds__(256) void final_residual_k(
    const float* __restrict__ outT, const float* __restrict__ x,
    const float* __restrict__ gamma, float* __restrict__ res, int C, int N) {
  __shared__ float tile[32][33];
  const int b = blockIdx.z;
  const int n0 = blockIdx.x * 32, c0 = blockIdx.y * 32;
  const int tx = threadIdx.x, ty = threadIdx.y;
  const float* oT = outT + (long long)b * N * C;
  for (int i = ty; i < 32; i += 8)
    tile[i][tx] = oT[(long long)(n0 + i) * C + (c0 + tx)];
  __syncthreads();
  const float g = gamma[0];
  const float* xb = x + (long long)b * C * N;
  float* rb = res + (long long)b * C * N;
  for (int i = ty; i < 32; i += 8) {
    const long long idx = (long long)(c0 + i) * N + (n0 + tx);
    rb[idx] = g * tile[tx][i] + xb[idx];
  }
}

extern "C" void kernel_launch(void* const* d_in, const int* in_sizes, int n_in,
                              void* d_out, int out_size, void* d_ws, size_t ws_size,
                              hipStream_t stream) {
  (void)in_sizes; (void)n_in; (void)out_size; (void)ws_size;
  const float* x     = (const float*)d_in[0];
  const float* y     = (const float*)d_in[1];
  const float* Wq    = (const float*)d_in[2];
  const float* bq    = (const float*)d_in[3];
  const float* Wk    = (const float*)d_in[4];
  const float* bk    = (const float*)d_in[5];
  const float* Wv    = (const float*)d_in[6];
  const float* bv    = (const float*)d_in[7];
  const float* gamma = (const float*)d_in[8];

  constexpr int B = 4, C = 512, N = 4096, D = 32;
  constexpr long long sXY = (long long)C * N;   // per-batch x/y/v stride
  constexpr long long sQK = (long long)N * D;   // per-batch q/k stride
  constexpr long long sE  = (long long)N * N;   // per-batch energy stride
  constexpr long long sO  = (long long)N * C;   // per-batch outT stride

  char* ws = (char*)d_ws;
  _Float16* qh   = (_Float16*)(ws);                                        //  1 MB
  _Float16* kh   = (_Float16*)(ws + ((size_t)1 << 20));                    //  1 MB
  _Float16* vh   = (_Float16*)(ws + ((size_t)2 << 20));                    // 16 MB
  _Float16* E    = (_Float16*)(ws + ((size_t)18 << 20));                   // 128 MB
  float*    outT = (float*)   (ws + ((size_t)146 << 20));                  // 32 MB

  // q[b][n][d] = sum_c x[b][c][n] * Wq[d][c] + bq[d]   (A = x^T via TA, Bt = Wq)
  gemm_wmma_k<128, 32, 16, 32, true, false, float, true, 1>
      <<<dim3(1, N / 128, B), dim3(256), 0, stream>>>(
          x, Wq, bq, qh, C, /*lda*/ N, /*ldb*/ C, /*ldd*/ D, sXY, 0, sQK);

  // k[b][n][d] from y, Wk
  gemm_wmma_k<128, 32, 16, 32, true, false, float, true, 1>
      <<<dim3(1, N / 128, B), dim3(256), 0, stream>>>(
          y, Wk, bk, kh, C, N, C, D, sXY, 0, sQK);

  // v[b][e][n] = sum_c Wv[e][c] * y[b][c][n] + bv[e]   (A = Wv, Bt = y^T via TB)
  gemm_wmma_k<64, 256, 32, 64, false, true, float, true, 2>
      <<<dim3(N / 256, C / 64, B), dim3(256), 0, stream>>>(
          Wv, y, bv, vh, C, /*lda*/ C, /*ldb*/ N, /*ldd*/ N, 0, sXY, sXY);

  // energy[b][i][j] = sum_d q[i][d] * k[j][d]   (A = qh, Bt = kh), f16 out
  gemm_wmma_k<64, 256, 32, 64, false, false, _Float16, true, 0>
      <<<dim3(N / 256, N / 64, B), dim3(256), 0, stream>>>(
          qh, kh, nullptr, E, D, D, D, N, sQK, sQK, sE);

  // softmax over each row of E, in place
  softmax_rows_k<<<dim3(B * N), dim3(256), 0, stream>>>(E);

  // outT[b][i][c] = sum_j att[i][j] * v[c][j]   (A = att, Bt = vh), f32 out
  gemm_wmma_k<64, 256, 32, 64, false, false, _Float16, false, 0>
      <<<dim3(C / 256, N / 64, B), dim3(256), 0, stream>>>(
          E, vh, nullptr, outT, N, N, N, C, sE, sXY, sO);

  // result = gamma * outT^T + x
  final_residual_k<<<dim3(N / 32, C / 32, B), dim3(32, 8), 0, stream>>>(
      outT, x, gamma, (float*)d_out, C, N);
}
